// Net_21225728377473
// MI455X (gfx1250) — matrix operand order repaired and checked by reading, not compiled, not run on hardware
//
#include <hip/hip_runtime.h>
#include <math.h>

#define NPTS 16384
#define NB   8
#define NP   2048
#define KNN  20
#define ASTR 40   // LDS row stride in halves for 32-wide K tiles: 80B rows, 16B aligned

typedef _Float16 half8  __attribute__((ext_vector_type(8)));
typedef _Float16 half16 __attribute__((ext_vector_type(16)));
typedef float    float8 __attribute__((ext_vector_type(8)));

// ---------------------------------------------------------------- helpers

__device__ inline half16 combine8(half8 lo, half8 hi) {
  return __builtin_shufflevector(lo, hi, 0,1,2,3,4,5,6,7,8,9,10,11,12,13,14,15);
}
// A fragment (16x32 f16): lane half h: k = h*8..+7 and 16+h*8..+7 of its row.
__device__ inline half16 frag_a_ptr(const _Float16* row, int h) {
  half8 lo = *(const half8*)(row + h * 8);
  half8 hi = *(const half8*)(row + 16 + h * 8);
  return combine8(lo, hi);
}
// B fragment (32x16 f16) stored [n][k]: lane half h: k = h*8..+7 and 16+h*8..+7 of its row.
__device__ inline half16 frag_b_ptr(const _Float16* row, int h) {
  half8 lo = *(const half8*)(row + h * 8);
  half8 hi = *(const half8*)(row + 16 + h * 8);
  return combine8(lo, hi);
}
__device__ inline float8 wmma_f16(half16 a, half16 b, float8 c) {
  return __builtin_amdgcn_wmma_f32_16x16x32_f16(false, a, false, b, (short)0, c, false, false);
}

// Async DMA: copy 16B from global to LDS, tracked by ASYNCcnt (bypasses VGPRs).
// Generic shared-pointer low 32 bits == LDS offset (aperture in the high dword).
__device__ inline unsigned lds_off_of(const void* p) {
  return (unsigned)(uintptr_t)p;
}
__device__ inline void async_cp16(unsigned lds_off, const _Float16* src) {
  asm volatile("global_load_async_to_lds_b128 %0, %1, off"
               :: "v"(lds_off), "v"(src) : "memory");
}
__device__ inline void wait_async6() { asm volatile("s_wait_asynccnt 0x6" ::: "memory"); }
__device__ inline void wait_async0() { asm volatile("s_wait_asynccnt 0x0" ::: "memory"); }

// ---------------------------------------------------------------- async GEMM
// Out[m,n] = bias[n] + sum_k A[m*lda+k] * W[n*ldw+k]  (A @ W^T), f16 inputs.
// 128 threads = 4 waves; tile 128x64 (wave 32x64); K-step 32; K % 32 == 0,
// M % 128 == 0, N % 64 == 0 (all call sites). Double-buffered async-to-LDS staging:
// 6 async b128 per wave per step; s_wait_asynccnt 6 retires step k while k+1 flies.
__global__ __launch_bounds__(128) void gemm_f16a(
    const _Float16* __restrict__ A, int lda,
    const _Float16* __restrict__ W, int ldw,
    const float* __restrict__ bias,
    void* __restrict__ OutP, int ldo,
    int M, int N, int Kd, int out_f16)
{
  __shared__ _Float16 As[2][128 * ASTR];
  __shared__ _Float16 Ws[2][64 * ASTR];
  const int t = threadIdx.x;
  const int lane = t & 31, w = t >> 5;
  const int m0 = blockIdx.y * 128, n0 = blockIdx.x * 64;
  const int h = lane >> 4, lm = lane & 15;
  const int srow = t >> 1, skb = (t & 1) * 16;
  const int nK = Kd >> 5;

  float8 acc[2][4];
#pragma unroll
  for (int mt = 0; mt < 2; ++mt)
#pragma unroll
    for (int nt = 0; nt < 4; ++nt)
#pragma unroll
      for (int r = 0; r < 8; ++r) acc[mt][nt][r] = 0.f;

  const _Float16* arow = A + (size_t)(m0 + t) * lda;          // thread stages its own A row
  const _Float16* wrow = W + (size_t)(n0 + srow) * ldw + skb; // two threads per W row

  auto issue = [&](int kt) {
    const int buf = kt & 1;
    const _Float16* as = arow + kt * 32;
#pragma unroll
    for (int c = 0; c < 4; ++c)
      async_cp16(lds_off_of(&As[buf][t * ASTR + c * 8]), as + c * 8);
    const _Float16* ws = wrow + kt * 32;
#pragma unroll
    for (int c = 0; c < 2; ++c)
      async_cp16(lds_off_of(&Ws[buf][srow * ASTR + skb + c * 8]), ws + c * 8);
  };

  issue(0);
  for (int kt = 0; kt < nK; ++kt) {
    const int cur = kt & 1;
    if (kt + 1 < nK) { issue(kt + 1); wait_async6(); } else { wait_async0(); }
    __syncthreads();                       // step-kt tiles resident for all waves
    half16 af0 = frag_a_ptr(&As[cur][(w * 32 + lm) * ASTR], h);
    half16 af1 = frag_a_ptr(&As[cur][(w * 32 + 16 + lm) * ASTR], h);
#pragma unroll
    for (int nt = 0; nt < 4; ++nt) {
      half16 bf = frag_b_ptr(&Ws[cur][(nt * 16 + lm) * ASTR], h);
      acc[0][nt] = wmma_f16(af0, bf, acc[0][nt]);
      acc[1][nt] = wmma_f16(af1, bf, acc[1][nt]);
    }
    __syncthreads();                       // readers done: buf[cur] reusable at kt+2
  }

  if (out_f16) {
    _Float16* o16 = (_Float16*)OutP;
#pragma unroll
    for (int mt = 0; mt < 2; ++mt)
#pragma unroll
      for (int nt = 0; nt < 4; ++nt) {
        int col = n0 + nt * 16 + lm;
        float bv = (bias != nullptr) ? bias[col] : 0.f;
#pragma unroll
        for (int r = 0; r < 8; ++r) {
          int row = m0 + w * 32 + mt * 16 + h * 8 + r;
          o16[(size_t)row * ldo + col] = (_Float16)(acc[mt][nt][r] + bv);
        }
      }
  } else {
    float* o32 = (float*)OutP;
#pragma unroll
    for (int mt = 0; mt < 2; ++mt)
#pragma unroll
      for (int nt = 0; nt < 4; ++nt) {
        int col = n0 + nt * 16 + lm;
        float bv = (bias != nullptr) ? bias[col] : 0.f;
#pragma unroll
        for (int r = 0; r < 8; ++r) {
          int row = m0 + w * 32 + mt * 16 + h * 8 + r;
          o32[(size_t)row * ldo + col] = acc[mt][nt][r] + bv;
        }
      }
  }
}

// ---------------------------------------------------------------- fc1 + per-batch max-pool
// Same async pipeline; A = concat(x1,x2,x3,feats) (boundaries 64/192/448, multiples of 32,
// so a K-step never straddles sources). Epilogue: LDS row-max + global_atomic_max_num_f32.
__global__ __launch_bounds__(128) void gemm_cat_max_a(
    const _Float16* __restrict__ X1, const _Float16* __restrict__ X2,
    const _Float16* __restrict__ X3, const _Float16* __restrict__ F,
    const _Float16* __restrict__ W, float* __restrict__ g)
{
  __shared__ _Float16 As[2][128 * ASTR];
  __shared__ _Float16 Ws[2][64 * ASTR];
  __shared__ float Cs[128 * 65];
  const int t = threadIdx.x;
  const int lane = t & 31, w = t >> 5;
  const int m0 = blockIdx.y * 128, n0 = blockIdx.x * 64;
  const int h = lane >> 4, lm = lane & 15;
  const int srow = t >> 1, skb = (t & 1) * 16;

  float8 acc[2][4];
#pragma unroll
  for (int mt = 0; mt < 2; ++mt)
#pragma unroll
    for (int nt = 0; nt < 4; ++nt)
#pragma unroll
      for (int r = 0; r < 8; ++r) acc[mt][nt][r] = 0.f;

  const _Float16* wrow = W + (size_t)(n0 + srow) * 576 + skb;

  auto issue = [&](int kt) {
    const int buf = kt & 1;
    const int kb = kt * 32;
    const _Float16* src; int sld, soff;
    if (kb < 64)        { src = X1; sld = 64;  soff = kb; }
    else if (kb < 192)  { src = X2; sld = 128; soff = kb - 64; }
    else if (kb < 448)  { src = X3; sld = 256; soff = kb - 192; }
    else                { src = F;  sld = 128; soff = kb - 448; }
    const _Float16* as = src + (size_t)(m0 + t) * sld + soff;
#pragma unroll
    for (int c = 0; c < 4; ++c)
      async_cp16(lds_off_of(&As[buf][t * ASTR + c * 8]), as + c * 8);
    const _Float16* ws = wrow + kb;
#pragma unroll
    for (int c = 0; c < 2; ++c)
      async_cp16(lds_off_of(&Ws[buf][srow * ASTR + skb + c * 8]), ws + c * 8);
  };

  issue(0);
  for (int kt = 0; kt < 18; ++kt) {        // K = 576 = 18*32
    const int cur = kt & 1;
    if (kt + 1 < 18) { issue(kt + 1); wait_async6(); } else { wait_async0(); }
    __syncthreads();
    half16 af0 = frag_a_ptr(&As[cur][(w * 32 + lm) * ASTR], h);
    half16 af1 = frag_a_ptr(&As[cur][(w * 32 + 16 + lm) * ASTR], h);
#pragma unroll
    for (int nt = 0; nt < 4; ++nt) {
      half16 bf = frag_b_ptr(&Ws[cur][(nt * 16 + lm) * ASTR], h);
      acc[0][nt] = wmma_f16(af0, bf, acc[0][nt]);
      acc[1][nt] = wmma_f16(af1, bf, acc[1][nt]);
    }
    __syncthreads();
  }

#pragma unroll
  for (int mt = 0; mt < 2; ++mt)
#pragma unroll
    for (int nt = 0; nt < 4; ++nt)
#pragma unroll
      for (int r = 0; r < 8; ++r)
        Cs[(w * 32 + mt * 16 + h * 8 + r) * 65 + nt * 16 + lm] = acc[mt][nt][r];
  __syncthreads();
  if (t < 64) {
    float mv = -3.0e38f;
    for (int r = 0; r < 128; ++r) mv = fmaxf(mv, Cs[r * 65 + t]);
    int batch = m0 / NP;
    float* gp = g + batch * 1024 + n0 + t;
    asm volatile("global_atomic_max_num_f32 %0, %1, off" :: "v"(gp), "v"(mv) : "memory");
  }
}

// ---------------------------------------------------------------- one-time converts

__global__ void cvt_f16_k(const float* __restrict__ src, _Float16* __restrict__ dst, int n4) {
  int i = blockIdx.x * blockDim.x + threadIdx.x;  // n4 = n/4, n % 4 == 0, 16B aligned
  if (i < n4) {
    float4 v = ((const float4*)src)[i];
    dst[i * 4 + 0] = (_Float16)v.x;
    dst[i * 4 + 1] = (_Float16)v.y;
    dst[i * 4 + 2] = (_Float16)v.z;
    dst[i * 4 + 3] = (_Float16)v.w;
  }
}

// weight slice -> f16 [O][Kw], zero-padded past D
__global__ void prep_w16_k(const float* __restrict__ W, int ldw, int woff,
                           int O, int D, int Kw, _Float16* __restrict__ out) {
  int e = blockIdx.x * blockDim.x + threadIdx.x;
  if (e < O * Kw) {
    int o = e / Kw, k = e - o * Kw;
    float v = (k < D) ? W[(size_t)o * ldw + woff + k] : 0.f;
    out[e] = (_Float16)v;
  }
}

// ---------------------------------------------------------------- EdgeConv edge-part + max
// One wave per point. Edge matrix (nb - x): 20 x Sx (f16, padded tail is 0-0=0) staged in
// LDS as WMMA B via half8 vector ops. A = pre-laid f16 Wb tiles (global b128, L2-resident).
// shfl-xor max over neighbors, add Ya, emit f16 next-layer features.
__global__ __launch_bounds__(128) void edgeconv_max(
    const _Float16* __restrict__ X, int Sx,
    const int* __restrict__ idx,
    const _Float16* __restrict__ Wb,  // [O][Sx] f16
    const float* __restrict__ Ya,
    _Float16* __restrict__ Out, int O)
{
  extern __shared__ _Float16 sm[];
  const int t = threadIdx.x, lane = t & 31, w = t >> 5;
  const int p = blockIdx.x * 4 + w;
  const int stride = Sx + 8;                 // 16B-aligned rows, staggered banks
  _Float16* Eb = sm + w * 32 * stride;
  const int nc = Sx >> 3;                    // half8 chunks per row

  {  // stage edge tile: row n = neighbor; lanes >= KNN zero-fill (one wave-level branch)
    const int n = lane;
    half8* dst = (half8*)(Eb + n * stride);
    if (n < KNN) {
      const half8* xp = (const half8*)(X + (size_t)p * Sx);
      const half8* xn = (const half8*)(X + (size_t)idx[p * KNN + n] * Sx);
      for (int c = 0; c < nc; ++c) dst[c] = xn[c] - xp[c];   // vector f16 subtract
    } else {
      half8 z = (half8)(_Float16)0.f;
      for (int c = 0; c < nc; ++c) dst[c] = z;
    }
  }
  __syncthreads();

  const int h = lane >> 4, lm = lane & 15;
  const int nK = Sx >> 5;
  const int nMt = O >> 4;

  for (int mt = 0; mt < nMt; ++mt) {
    float8 a0, a1;
#pragma unroll
    for (int r = 0; r < 8; ++r) { a0[r] = 0.f; a1[r] = 0.f; }
    const _Float16* wrow = Wb + (size_t)(mt * 16 + lm) * Sx;
    for (int kt = 0; kt < nK; ++kt) {
      half16 af = frag_a_ptr(wrow + kt * 32, h);
      half16 b0 = frag_b_ptr(&Eb[lm * stride + kt * 32], h);
      half16 b1 = frag_b_ptr(&Eb[(16 + lm) * stride + kt * 32], h);
      a0 = wmma_f16(af, b0, a0);
      a1 = wmma_f16(af, b1, a1);
    }
#pragma unroll
    for (int r = 0; r < 8; ++r) {
      float cand = fmaxf(a0[r], (lm < 4) ? a1[r] : -3.0e38f);
      cand = fmaxf(cand, __shfl_xor(cand, 8, 32));
      cand = fmaxf(cand, __shfl_xor(cand, 4, 32));
      cand = fmaxf(cand, __shfl_xor(cand, 2, 32));
      cand = fmaxf(cand, __shfl_xor(cand, 1, 32));
      if (lm == 0) {
        int o = mt * 16 + h * 8 + r;
        Out[(size_t)p * O + o] = (_Float16)(Ya[(size_t)p * O + o] + cand);
      }
    }
  }
}

// ---------------------------------------------------------------- small kernels

__global__ void sqnorm_k(const _Float16* __restrict__ X, int Sx, float* __restrict__ sq) {
  int i = blockIdx.x * blockDim.x + threadIdx.x;
  if (i < NPTS) {
    const half8* xp = (const half8*)(X + (size_t)i * Sx);
    float s = 0.f;
    for (int c = 0; c < (Sx >> 3); ++c) {
      half8 v = xp[c];
#pragma unroll
      for (int j = 0; j < 8; ++j) { float f = (float)v[j]; s += f * f; }
    }
    sq[i] = s;
  }
}

// top-20 nearest (incl. self); dist(i,j) ~ sq[j] - 2*G[i,j]; G symmetric -> read transposed.
__global__ void topk20_k(const float* __restrict__ G, const float* __restrict__ sq,
                         int b, int* __restrict__ idx) {
  int i = blockIdx.x * blockDim.x + threadIdx.x;
  if (i >= NP) return;
  float best[KNN]; int bi[KNN];
#pragma unroll
  for (int q = 0; q < KNN; ++q) { best[q] = 3.0e38f; bi[q] = 0; }
  const float* sb = sq + b * NP;
  for (int j = 0; j < NP; ++j) {
    float d = sb[j] - 2.0f * G[(size_t)j * NP + i];
    if (d < best[KNN - 1]) {
      best[KNN - 1] = d; bi[KNN - 1] = j;
#pragma unroll
      for (int q = KNN - 1; q >= 1; --q) {
        if (best[q] < best[q - 1]) {
          float tv = best[q]; best[q] = best[q - 1]; best[q - 1] = tv;
          int ti = bi[q]; bi[q] = bi[q - 1]; bi[q - 1] = ti;
        }
      }
    }
  }
  int* op = idx + (size_t)(b * NP + i) * KNN;
#pragma unroll
  for (int q = 0; q < KNN; ++q) op[q] = b * NP + bi[q];
}

// h0 = [pos | x | feats] padded to 160 f16 (zero tail keeps all GEMM K-dims 32-aligned)
__global__ void build_h0_k(const float* __restrict__ pos, const float* __restrict__ x,
                           const _Float16* __restrict__ F, _Float16* __restrict__ H0) {
  int e = blockIdx.x * blockDim.x + threadIdx.x;
  if (e < NPTS * 160) {
    int i = e / 160, c = e - 160 * i;
    float v = (c < 3) ? pos[i * 3 + c]
            : (c < 6) ? x[i * 3 + (c - 3)]
            : (c < 134) ? (float)F[(size_t)i * 128 + (c - 6)]
            : 0.f;
    H0[e] = (_Float16)v;
  }
}

__global__ void init_g_k(float* g) {
  int e = blockIdx.x * blockDim.x + threadIdx.x;
  if (e < NB * 1024) g[e] = -3.0e38f;
}

// head: g+bfc1 -> 512 relu -> 256 relu -> 23 -> log_softmax (tiny; one block)
__global__ __launch_bounds__(256) void head_mlp_k(
    const float* __restrict__ g, const float* __restrict__ bfc1,
    const float* __restrict__ Wfa, const float* __restrict__ bfa,
    const float* __restrict__ Wfb, const float* __restrict__ bfb,
    const float* __restrict__ Wfc, const float* __restrict__ bfc,
    float* __restrict__ out)
{
  __shared__ float gf[NB * 1024];
  __shared__ float a1[NB * 512];
  __shared__ float a2[NB * 256];
  __shared__ float lg[NB * 24];
  const int t = threadIdx.x;
  for (int e = t; e < NB * 1024; e += 256) gf[e] = g[e] + bfc1[e & 1023];
  __syncthreads();
  for (int e = t; e < NB * 512; e += 256) {
    int b = e >> 9, o = e & 511;
    float s = bfa[o];
    const float* wr = Wfa + (size_t)o * 1024;
    for (int k = 0; k < 1024; ++k) s += gf[(b << 10) + k] * wr[k];
    a1[e] = fmaxf(s, 0.f);
  }
  __syncthreads();
  for (int e = t; e < NB * 256; e += 256) {
    int b = e >> 8, o = e & 255;
    float s = bfb[o];
    const float* wr = Wfb + (size_t)o * 512;
    for (int k = 0; k < 512; ++k) s += a1[(b << 9) + k] * wr[k];
    a2[e] = fmaxf(s, 0.f);
  }
  __syncthreads();
  for (int e = t; e < NB * 23; e += 256) {
    int b = e / 23, o = e - 23 * b;
    float s = bfc[o];
    const float* wr = Wfc + (size_t)o * 256;
    for (int k = 0; k < 256; ++k) s += a2[(b << 8) + k] * wr[k];
    lg[b * 24 + o] = s;
  }
  __syncthreads();
  if (t < NB) {
    float m = -3.0e38f;
    for (int o = 0; o < 23; ++o) m = fmaxf(m, lg[t * 24 + o]);
    float s = 0.f;
    for (int o = 0; o < 23; ++o) s += expf(lg[t * 24 + o] - m);
    float ls = logf(s);
    for (int o = 0; o < 23; ++o) out[t * 23 + o] = lg[t * 24 + o] - m - ls;
  }
}

// ---------------------------------------------------------------- launch

extern "C" void kernel_launch(void* const* d_in, const int* in_sizes, int n_in,
                              void* d_out, int out_size, void* d_ws, size_t ws_size,
                              hipStream_t stream) {
  const float* pos      = (const float*)d_in[0];
  const float* xin      = (const float*)d_in[1];
  const float* features = (const float*)d_in[2];
  // d_in[3] = batch (implied by layout, unused)
  const float* Wf   = (const float*)d_in[4];
  const float* bf   = (const float*)d_in[5];
  const float* W1   = (const float*)d_in[6];
  const float* b1   = (const float*)d_in[7];
  const float* W2   = (const float*)d_in[8];
  const float* b2   = (const float*)d_in[9];
  const float* W3   = (const float*)d_in[10];
  const float* b3   = (const float*)d_in[11];
  const float* Wfc1 = (const float*)d_in[12];
  const float* bfc1 = (const float*)d_in[13];
  const float* Wfa  = (const float*)d_in[14];
  const float* bfa  = (const float*)d_in[15];
  const float* Wfb  = (const float*)d_in[16];
  const float* bfb  = (const float*)d_in[17];
  const float* Wfc  = (const float*)d_in[18];
  const float* bfc  = (const float*)d_in[19];
  float* out = (float*)d_out;
  (void)in_sizes; (void)n_in; (void)out_size; (void)ws_size;

  char* p = (char*)d_ws;
  auto carve = [&](size_t bytes) -> void* {
    void* r = (void*)p; p += (bytes + 255) & ~(size_t)255; return r;
  };
  _Float16* FEAT16 = (_Float16*)carve((size_t)NPTS * 1344 * 2);
  _Float16* WF16   = (_Float16*)carve((size_t)128 * 1344 * 2);
  _Float16* WFC116 = (_Float16*)carve((size_t)1024 * 576 * 2);
  _Float16* WA16   = (_Float16*)carve((size_t)256 * 160 * 2);
  _Float16* WB16   = (_Float16*)carve((size_t)256 * 160 * 2);
  _Float16* F      = (_Float16*)carve((size_t)NPTS * 128 * 2);
  _Float16* H0     = (_Float16*)carve((size_t)NPTS * 160 * 2);
  _Float16* X1     = (_Float16*)carve((size_t)NPTS * 64 * 2);
  _Float16* X2     = (_Float16*)carve((size_t)NPTS * 128 * 2);
  _Float16* X3     = (_Float16*)carve((size_t)NPTS * 256 * 2);
  float*    Ya     = (float*)carve((size_t)NPTS * 256 * 4);
  float*    SQ     = (float*)carve((size_t)NPTS * 4);
  float*    GR     = (float*)carve((size_t)NP * NP * 4);
  int*      IDX    = (int*)carve((size_t)NPTS * KNN * 4);
  float*    G      = (float*)carve((size_t)NB * 1024 * 4);

  dim3 blk(128);

  // one-time f16 conversions / weight layouts
  cvt_f16_k<<<(NPTS * 1344 / 4 + 255) / 256, 256, 0, stream>>>(features, FEAT16, NPTS * 1344 / 4);
  prep_w16_k<<<(128 * 1344 + 255) / 256, 256, 0, stream>>>(Wf, 1344, 0, 128, 1344, 1344, WF16);
  prep_w16_k<<<(1024 * 576 + 255) / 256, 256, 0, stream>>>(Wfc1, 576, 0, 1024, 576, 576, WFC116);

  // feats = features @ Wf^T + bf   (f16 out)
  gemm_f16a<<<dim3(128 / 64, NPTS / 128), blk, 0, stream>>>(
      FEAT16, 1344, WF16, 1344, bf, F, 128, NPTS, 128, 1344, 1);
  build_h0_k<<<(NPTS * 160 + 255) / 256, 256, 0, stream>>>(pos, xin, F, H0);

  struct Layer { const _Float16* Xin; int Sx; int D; const float* W; int ldw; const float* b; int O; _Float16* Xout; };
  Layer L[3] = {
    { H0, 160, 134, W1, 268, b1,  64, X1 },
    { X1,  64,  64, W2, 128, b2, 128, X2 },
    { X2, 128, 128, W3, 256, b3, 256, X3 },
  };
  for (int li = 0; li < 3; ++li) {
    const Layer& ly = L[li];
    sqnorm_k<<<(NPTS + 255) / 256, 256, 0, stream>>>(ly.Xin, ly.Sx, SQ);
    for (int b = 0; b < NB; ++b) {
      const _Float16* Xb = ly.Xin + (size_t)b * NP * ly.Sx;
      // Gram = Xb @ Xb^T (zero-padded K is inert), f32 out
      gemm_f16a<<<dim3(NP / 64, NP / 128), blk, 0, stream>>>(
          Xb, ly.Sx, Xb, ly.Sx, nullptr, GR, NP, NP, NP, ly.Sx, 0);
      topk20_k<<<NP / 256, 256, 0, stream>>>(GR, SQ, b, IDX);
    }
    // weight slices -> padded f16
    prep_w16_k<<<(ly.O * ly.Sx + 255) / 256, 256, 0, stream>>>(
        ly.W, ly.ldw, 0, ly.O, ly.D, ly.Sx, WA16);
    prep_w16_k<<<(ly.O * ly.Sx + 255) / 256, 256, 0, stream>>>(
        ly.W, ly.ldw, ly.D, ly.O, ly.D, ly.Sx, WB16);
    // Ya = X @ Wa^T + b  (f32 out)
    gemm_f16a<<<dim3(ly.O / 64, NPTS / 128), blk, 0, stream>>>(
        ly.Xin, ly.Sx, WA16, ly.Sx, ly.b, Ya, ly.O, NPTS, ly.O, ly.Sx, 0);
    size_t sh = (size_t)4 * 32 * (ly.Sx + 8) * sizeof(_Float16);
    edgeconv_max<<<NPTS / 4, blk, sh, stream>>>(
        ly.Xin, ly.Sx, IDX, WB16, Ya, ly.Xout, ly.O);
  }

  init_g_k<<<(NB * 1024 + 255) / 256, 256, 0, stream>>>(G);
  gemm_cat_max_a<<<dim3(1024 / 64, NPTS / 128), blk, 0, stream>>>(X1, X2, X3, F, WFC116, G);
  head_mlp_k<<<1, 256, 0, stream>>>(G, bfc1, Wfa, bfa, Wfb, bfb, Wfc, bfc, out);
}
